// MemDecoder_23519240913098
// MI455X (gfx1250) — compile-verified, matrix operand order checked by reading, not run on hardware
//
#include <hip/hip_runtime.h>
#include <hip/hip_bf16.h>

typedef __attribute__((ext_vector_type(16))) _Float16 v16h;
typedef __attribute__((ext_vector_type(8)))  float    v8f;

#define NW    16     // waves per block (512 threads)
#define MTILE 64
#define LDA   1032   // act row stride in halves (2064B: 16B-aligned, off the 256B bank period)
#define LDT   520    // tmp row stride in halves (1040B)

// ---------------------------------------------------------------------------
// fragment helpers
// ---------------------------------------------------------------------------
// row-major source (LDS activations): chunks at +0 and +16 halves
__device__ inline v16h load_frag(const _Float16* p) {
  union { v16h v; uint4 q[2]; } f;
  f.q[0] = *(const uint4*)(p);
  f.q[1] = *(const uint4*)(p + 16);
  return f.v;
}
// fragment-packed source (weights/grid in global): 32B contiguous per lane
__device__ inline v16h load_fragP(const _Float16* p) {
  union { v16h v; uint4 q[2]; } f;
  f.q[0] = *(const uint4*)(p);
  f.q[1] = *(const uint4*)(p + 8);
  return f.v;
}

__device__ inline float lrelu(float v) { return v > 0.f ? v : 0.01f * v; }

// packed index (in halves) of element (n,k) of an [N][K] fragment-packed matrix:
// 1KB blocks, one per 16x32 tile; lane = (n&15)+16*((k>>3)&1); chunk = (k>>4)&1
__device__ inline size_t packed_idx(int n, int k, int K) {
  int bi = (n >> 4) * (K >> 5) + (k >> 5);
  int kk = k & 31;
  int hi = (kk >> 3) & 1;
  int ch = (kk >> 4) & 1;
  int ln = (n & 15) + (hi << 4);
  return ((size_t)bi << 9) + (ln << 4) + (ch << 3) + (kk & 7);
}

// ---------------------------------------------------------------------------
// conv stack (tiny): direct 3x3 grouped conv + leaky relu
// ---------------------------------------------------------------------------
__global__ void conv3x3_lrelu(const float* __restrict__ in, const float* __restrict__ w,
                              const float* __restrict__ b, float* __restrict__ out,
                              int Cin, int Cout, int inH, int inW, int groups, int pad) {
  int outH = pad ? inH : inH - 2;
  int outW = pad ? inW : inW - 2;
  int idx = blockIdx.x * blockDim.x + threadIdx.x;
  int total = Cout * outH * outW;
  if (idx >= total) return;
  int x = idx % outW;
  int y = (idx / outW) % outH;
  int co = idx / (outW * outH);
  int cig = Cin / groups, cog = Cout / groups;
  int g = co / cog;
  const float* wp = w + (size_t)co * cig * 9;
  float acc = b[co];
  for (int ci = 0; ci < cig; ++ci) {
    const float* ip = in + (size_t)(g * cig + ci) * inH * inW;
    const float* wc = wp + ci * 9;
    for (int ky = 0; ky < 3; ++ky) {
      int iy = y + ky - pad;
      if (iy < 0 || iy >= inH) continue;
      for (int kx = 0; kx < 3; ++kx) {
        int ix = x + kx - pad;
        if (ix < 0 || ix >= inW) continue;
        acc += ip[iy * inW + ix] * wc[ky * 3 + kx];
      }
    }
  }
  out[idx] = lrelu(acc);
}

// ---------------------------------------------------------------------------
// positional grid -> f16, fragment-packed as a [65536][32] A-source
// ---------------------------------------------------------------------------
__global__ void make_grid(_Float16* __restrict__ g) {
  int idx = blockIdx.x * blockDim.x + threadIdx.x;   // 65536*32
  int c = idx & 31, p = idx >> 5;
  int i = p >> 8, j = p & 255;
  float r = (float)(((c & 1) ? j : i) - 128) * (3.14159265358979323846f / 256.0f);
  float scale = exp2f(0.5f * (float)(c >> 1));
  g[packed_idx(p, c, 32)] = (_Float16)sinf(r * scale);
}

// ---------------------------------------------------------------------------
// weight norm: w = g * v / ||v|| (per row), output f16 fragment-packed per layer
// rows = layers*N; each layer is an independent [N][K] packed matrix
// ---------------------------------------------------------------------------
__global__ void weight_norm_pack(const float* __restrict__ v, const float* __restrict__ g,
                                 _Float16* __restrict__ out, int rows, int N, int K) {
  int r = blockIdx.x * blockDim.x + threadIdx.x;
  if (r >= rows) return;
  const float* vr = v + (size_t)r * K;
  float ss = 0.f;
  for (int k = 0; k < K; ++k) ss += vr[k] * vr[k];
  float s = g[r] * rsqrtf(ss);
  int layer = r / N, n = r - layer * N;
  _Float16* o = out + (size_t)layer * N * K;
  for (int k = 0; k < K; ++k) o[packed_idx(n, k, K)] = (_Float16)(vr[k] * s);
}

// c1[n] = b1[n] + sum_k feat[k] * w1[n][32+k]   (B=1: feat part is a constant)
__global__ void feat_bias(const float* __restrict__ w1, const float* __restrict__ b1,
                          const float* __restrict__ feat, float* __restrict__ c1) {
  int n = blockIdx.x * blockDim.x + threadIdx.x;
  if (n >= 1024) return;
  const float* wr = w1 + (size_t)n * 2080 + 32;
  float acc = b1[n];
  for (int k = 0; k < 2048; ++k) acc += wr[k] * feat[k];
  c1[n] = acc;
}

// slice of a row-major f32 matrix -> f16 fragment-packed [N][K]
__global__ void slice_pack_f16(const float* __restrict__ src, _Float16* __restrict__ dst,
                               int N, int srcLd, int srcOff, int K) {
  int idx = blockIdx.x * blockDim.x + threadIdx.x;
  if (idx >= N * K) return;
  int n = idx / K, k = idx - n * K;
  dst[packed_idx(n, k, K)] = (_Float16)src[(size_t)n * srcLd + srcOff + k];
}

// ---------------------------------------------------------------------------
// fused per-pixel MLP: all 17 layers resident in LDS per 64-pixel tile.
// Per wave: m0 fixed => A K-panel preloaded once into registers per phase.
// ---------------------------------------------------------------------------
__global__ void __launch_bounds__(512) fused_mlp(
    const _Float16* __restrict__ grid16,          // packed [65536][32]
    const float*    __restrict__ c1,              // [1024]
    const _Float16* __restrict__ W1g,             // packed [1024][32]
    const _Float16* __restrict__ S1WF,            // packed [8][512][512]
    const _Float16* __restrict__ S1WG,
    const float*    __restrict__ S1BF,            // [8][512]
    const float*    __restrict__ S1BG,
    const _Float16* __restrict__ W2g,             // packed [512][32]
    const _Float16* __restrict__ W2m,             // packed [512][1024]
    const float*    __restrict__ b2,              // [512]
    const _Float16* __restrict__ S2WF,            // packed [8][256][256]
    const _Float16* __restrict__ S2WG,
    const float*    __restrict__ S2BF,            // [8][256]
    const float*    __restrict__ S2BG,
    const float*    __restrict__ w3,              // [3][512]
    const float*    __restrict__ b3,              // [3]
    float*          __restrict__ out)             // [3][65536]
{
  extern __shared__ __align__(16) _Float16 smem[];
  _Float16* act = smem;               // [64][LDA], width 1024 used
  _Float16* tmp = smem + MTILE * LDA; // [64][LDT], width 512 used

  const int tid  = threadIdx.x;
  const int lane = tid & 31;
  const int wave = tid >> 5;
  const int pixBase = blockIdx.x * MTILE;
  const int l15  = lane & 15;
  const int koff = (lane >> 4) << 3;    // A-chunk select (row-major sources)
  const int rb0  = (lane >> 4) << 3;    // C-tile row sub-base
  const int m0   = (wave & 3) << 4;     // fixed 16-row slice per wave
  const int ntStart = wave >> 2;        // n-tile start, stride NW/4 = 4

  // ---- layer 1: act = lrelu(grid @ W1g^T + c1), [64 x 1024], K=32 ----
  {
    v16h a = load_fragP(grid16 + (((size_t)(pixBase + m0) >> 4) << 9) + (lane << 4));
    for (int nt = ntStart; nt < 64; nt += NW / 4) {
      int n0 = nt << 4;
      v16h b = load_fragP(W1g + ((size_t)nt << 9) + (lane << 4));
      v8f c = {};
      c = __builtin_amdgcn_wmma_f32_16x16x32_f16(false, a, false, b, (short)0, c, false, false);
      int col = n0 + l15;
      float bias = c1[col];
      int rb = m0 + rb0;
      #pragma unroll
      for (int r = 0; r < 8; ++r)
        act[(rb + r) * LDA + col] = (_Float16)lrelu(c[r] + bias);
    }
  }
  __syncthreads();

  // ---- coupling stack 1: 8 steps of 512-wide F/G, in place in LDS ----
  for (int s = 0; s < 8; ++s) {
    const _Float16* Wf = S1WF + (size_t)s * 512 * 512;
    const _Float16* Wg = S1WG + (size_t)s * 512 * 512;
    const float* bf = S1BF + s * 512;
    const float* bg = S1BG + s * 512;

    // y1 = h1 + lrelu(h2 @ Wf^T + bf): A = cols [512,1024), write cols [0,512)
    {
      v16h af[16];
      const _Float16* Arow = act + (size_t)(m0 + l15) * LDA + 512 + koff;
      #pragma unroll
      for (int kk = 0; kk < 16; ++kk) af[kk] = load_frag(Arow + kk * 32);
      for (int nt = ntStart; nt < 32; nt += NW / 4) {
        int n0 = nt << 4;
        const _Float16* Wb = Wf + ((size_t)nt << 13) + (lane << 4);  // nt*16blk*512
        v8f c = {};
        #pragma unroll
        for (int kk = 0; kk < 16; ++kk) {
          v16h b = load_fragP(Wb + ((size_t)kk << 9));
          c = __builtin_amdgcn_wmma_f32_16x16x32_f16(false, af[kk], false, b, (short)0, c, false, false);
        }
        int col = n0 + l15;
        float bias = bf[col];
        int rb = m0 + rb0;
        #pragma unroll
        for (int r = 0; r < 8; ++r) {
          int off = (rb + r) * LDA + col;
          act[off] = (_Float16)((float)act[off] + lrelu(c[r] + bias));
        }
      }
    }
    __syncthreads();

    // y2 = h2 + lrelu(y1 @ Wg^T + bg): A = cols [0,512), write cols [512,1024)
    {
      v16h af[16];
      const _Float16* Arow = act + (size_t)(m0 + l15) * LDA + koff;
      #pragma unroll
      for (int kk = 0; kk < 16; ++kk) af[kk] = load_frag(Arow + kk * 32);
      for (int nt = ntStart; nt < 32; nt += NW / 4) {
        int n0 = nt << 4;
        const _Float16* Wb = Wg + ((size_t)nt << 13) + (lane << 4);
        v8f c = {};
        #pragma unroll
        for (int kk = 0; kk < 16; ++kk) {
          v16h b = load_fragP(Wb + ((size_t)kk << 9));
          c = __builtin_amdgcn_wmma_f32_16x16x32_f16(false, af[kk], false, b, (short)0, c, false, false);
        }
        int col = n0 + l15;
        float bias = bg[col];
        int rb = m0 + rb0;
        #pragma unroll
        for (int r = 0; r < 8; ++r) {
          int off = (rb + r) * LDA + 512 + col;
          act[off] = (_Float16)((float)act[off] + lrelu(c[r] + bias));
        }
      }
    }
    __syncthreads();
  }

  // ---- layer w2: tmp = lrelu(mid @ W2m^T + grid @ W2g^T + b2), K=1024(+32) ----
  {
    const _Float16* Arow = act + (size_t)(m0 + l15) * LDA + koff;
    v16h ag = load_fragP(grid16 + (((size_t)(pixBase + m0) >> 4) << 9) + (lane << 4));
    for (int nt = ntStart; nt < 32; nt += NW / 4) {
      int n0 = nt << 4;
      const _Float16* Wb = W2m + ((size_t)nt << 14) + (lane << 4);   // nt*32blk*512
      v8f c = {};
      #pragma unroll 8
      for (int kk = 0; kk < 32; ++kk) {
        v16h a = load_frag(Arow + kk * 32);
        v16h b = load_fragP(Wb + ((size_t)kk << 9));
        c = __builtin_amdgcn_wmma_f32_16x16x32_f16(false, a, false, b, (short)0, c, false, false);
      }
      {
        v16h b = load_fragP(W2g + ((size_t)nt << 9) + (lane << 4));
        c = __builtin_amdgcn_wmma_f32_16x16x32_f16(false, ag, false, b, (short)0, c, false, false);
      }
      int col = n0 + l15;
      float bias = b2[col];
      int rb = m0 + rb0;
      #pragma unroll
      for (int r = 0; r < 8; ++r)
        tmp[(rb + r) * LDT + col] = (_Float16)lrelu(c[r] + bias);
    }
  }
  __syncthreads();

  // ---- coupling stack 2: 8 steps of 256-wide F/G, in place in tmp ----
  for (int s = 0; s < 8; ++s) {
    const _Float16* Wf = S2WF + (size_t)s * 256 * 256;
    const _Float16* Wg = S2WG + (size_t)s * 256 * 256;
    const float* bf = S2BF + s * 256;
    const float* bg = S2BG + s * 256;

    {
      v16h af[8];
      const _Float16* Arow = tmp + (size_t)(m0 + l15) * LDT + 256 + koff;
      #pragma unroll
      for (int kk = 0; kk < 8; ++kk) af[kk] = load_frag(Arow + kk * 32);
      for (int nt = ntStart; nt < 16; nt += NW / 4) {
        int n0 = nt << 4;
        const _Float16* Wb = Wf + ((size_t)nt << 12) + (lane << 4);  // nt*8blk*512
        v8f c = {};
        #pragma unroll
        for (int kk = 0; kk < 8; ++kk) {
          v16h b = load_fragP(Wb + ((size_t)kk << 9));
          c = __builtin_amdgcn_wmma_f32_16x16x32_f16(false, af[kk], false, b, (short)0, c, false, false);
        }
        int col = n0 + l15;
        float bias = bf[col];
        int rb = m0 + rb0;
        #pragma unroll
        for (int r = 0; r < 8; ++r) {
          int off = (rb + r) * LDT + col;
          tmp[off] = (_Float16)((float)tmp[off] + lrelu(c[r] + bias));
        }
      }
    }
    __syncthreads();

    {
      v16h af[8];
      const _Float16* Arow = tmp + (size_t)(m0 + l15) * LDT + koff;
      #pragma unroll
      for (int kk = 0; kk < 8; ++kk) af[kk] = load_frag(Arow + kk * 32);
      for (int nt = ntStart; nt < 16; nt += NW / 4) {
        int n0 = nt << 4;
        const _Float16* Wb = Wg + ((size_t)nt << 12) + (lane << 4);
        v8f c = {};
        #pragma unroll
        for (int kk = 0; kk < 8; ++kk) {
          v16h b = load_fragP(Wb + ((size_t)kk << 9));
          c = __builtin_amdgcn_wmma_f32_16x16x32_f16(false, af[kk], false, b, (short)0, c, false, false);
        }
        int col = n0 + l15;
        float bias = bg[col];
        int rb = m0 + rb0;
        #pragma unroll
        for (int r = 0; r < 8; ++r) {
          int off = (rb + r) * LDT + 256 + col;
          tmp[off] = (_Float16)((float)tmp[off] + lrelu(c[r] + bias));
        }
      }
    }
    __syncthreads();
  }

  // ---- head: rgb = sigmoid(tmp @ w3^T + b3) * 1.1 - 0.05, out [3][65536] ----
  if (tid < MTILE * 3) {
    int row = tid / 3, ch = tid - row * 3;
    const float* wr = w3 + ch * 512;
    const _Float16* ar = tmp + row * LDT;
    float acc = b3[ch];
    for (int k = 0; k < 512; ++k) acc += wr[k] * (float)ar[k];
    float sg = 1.f / (1.f + expf(-acc));
    out[(size_t)ch * 65536 + pixBase + row] = sg * 1.1f - 0.05f;
  }
}

// ---------------------------------------------------------------------------
// launch
// ---------------------------------------------------------------------------
extern "C" void kernel_launch(void* const* d_in, const int* in_sizes, int n_in,
                              void* d_out, int out_size, void* d_ws, size_t ws_size,
                              hipStream_t stream) {
  const float* feature = (const float*)d_in[0];
  const float* cw1 = (const float*)d_in[1];  const float* cb1 = (const float*)d_in[2];
  const float* cw2 = (const float*)d_in[3];  const float* cb2 = (const float*)d_in[4];
  const float* cw3 = (const float*)d_in[5];  const float* cb3 = (const float*)d_in[6];
  const float* cw4 = (const float*)d_in[7];  const float* cb4 = (const float*)d_in[8];
  const float* cw5 = (const float*)d_in[9];  const float* cb5 = (const float*)d_in[10];
  const float* w1  = (const float*)d_in[11]; const float* b1  = (const float*)d_in[12];
  const float* m1_vf = (const float*)d_in[13]; const float* m1_gf = (const float*)d_in[14];
  const float* m1_bf = (const float*)d_in[15];
  const float* m1_vg = (const float*)d_in[16]; const float* m1_gg = (const float*)d_in[17];
  const float* m1_bg = (const float*)d_in[18];
  const float* w2  = (const float*)d_in[19]; const float* b2  = (const float*)d_in[20];
  const float* m2_vf = (const float*)d_in[21]; const float* m2_gf = (const float*)d_in[22];
  const float* m2_bf = (const float*)d_in[23];
  const float* m2_vg = (const float*)d_in[24]; const float* m2_gg = (const float*)d_in[25];
  const float* m2_bg = (const float*)d_in[26];
  const float* w3  = (const float*)d_in[27]; const float* b3  = (const float*)d_in[28];
  float* out = (float*)d_out;

  char* ws = (char*)d_ws;
  size_t o = 0;
  float*    convA  = (float*)(ws + o);    o += (size_t)768 * 64 * 4;
  float*    convB  = (float*)(ws + o);    o += (size_t)768 * 64 * 4;
  float*    feat   = (float*)(ws + o);    o += (size_t)128 * 16 * 4;
  _Float16* grid16 = (_Float16*)(ws + o); o += (size_t)65536 * 32 * 2;
  _Float16* W1g    = (_Float16*)(ws + o); o += (size_t)1024 * 32 * 2;
  float*    c1     = (float*)(ws + o);    o += (size_t)1024 * 4;
  _Float16* S1WF   = (_Float16*)(ws + o); o += (size_t)8 * 512 * 512 * 2;
  _Float16* S1WG   = (_Float16*)(ws + o); o += (size_t)8 * 512 * 512 * 2;
  _Float16* W2g    = (_Float16*)(ws + o); o += (size_t)512 * 32 * 2;
  _Float16* W2m    = (_Float16*)(ws + o); o += (size_t)512 * 1024 * 2;
  _Float16* S2WF   = (_Float16*)(ws + o); o += (size_t)8 * 256 * 256 * 2;
  _Float16* S2WG   = (_Float16*)(ws + o); o += (size_t)8 * 256 * 256 * 2;

  // conv frontend (tiny)
  conv3x3_lrelu<<<(768 * 64 + 255) / 256, 256, 0, stream>>>(feature, cw1, cb1, convA, 1792, 768, 8, 8, 4, 1);
  conv3x3_lrelu<<<(768 * 64 + 255) / 256, 256, 0, stream>>>(convA, cw2, cb2, convB, 768, 768, 8, 8, 3, 1);
  conv3x3_lrelu<<<(768 * 64 + 255) / 256, 256, 0, stream>>>(convB, cw3, cb3, convA, 768, 768, 8, 8, 2, 1);
  conv3x3_lrelu<<<(768 * 36 + 255) / 256, 256, 0, stream>>>(convA, cw4, cb4, convB, 768, 768, 8, 8, 3, 0);
  conv3x3_lrelu<<<(128 * 16 + 255) / 256, 256, 0, stream>>>(convB, cw5, cb5, feat, 768, 128, 6, 6, 1, 0);

  // prep (all outputs fragment-packed f16)
  make_grid<<<(65536 * 32) / 256, 256, 0, stream>>>(grid16);
  weight_norm_pack<<<(4096 + 255) / 256, 256, 0, stream>>>(m1_vf, m1_gf, S1WF, 4096, 512, 512);
  weight_norm_pack<<<(4096 + 255) / 256, 256, 0, stream>>>(m1_vg, m1_gg, S1WG, 4096, 512, 512);
  weight_norm_pack<<<(2048 + 255) / 256, 256, 0, stream>>>(m2_vf, m2_gf, S2WF, 2048, 256, 256);
  weight_norm_pack<<<(2048 + 255) / 256, 256, 0, stream>>>(m2_vg, m2_gg, S2WG, 2048, 256, 256);
  feat_bias<<<4, 256, 0, stream>>>(w1, b1, feat, c1);
  slice_pack_f16<<<(1024 * 32 + 255) / 256, 256, 0, stream>>>(w1, W1g, 1024, 2080, 0, 32);
  slice_pack_f16<<<(512 * 32 + 255) / 256, 256, 0, stream>>>(w2, W2g, 512, 1056, 0, 32);
  slice_pack_f16<<<(512 * 1024 + 255) / 256, 256, 0, stream>>>(w2, W2m, 512, 1056, 32, 1024);

  // fused MLP: 1024 blocks x 512 threads; LDS = act[64][1032] + tmp[64][520] f16
  size_t smem = (size_t)(MTILE * LDA + MTILE * LDT) * sizeof(_Float16); // 198656 B
  fused_mlp<<<65536 / MTILE, 512, smem, stream>>>(
      grid16, c1, W1g, S1WF, S1WG, m1_bf, m1_bg,
      W2g, W2m, b2, S2WF, S2WG, m2_bf, m2_bg, w3, b3, out);
}
